// AttentionLayers_81509889343707
// MI455X (gfx1250) — matrix operand order repaired and checked by reading, not compiled, not run on hardware
//
#include <hip/hip_runtime.h>
#include <hip/hip_bf16.h>
#include <math.h>

// ---------------------------------------------------------------------------
// MHA block for MI455X (gfx1250): LayerNorm -> QKV proj -> flash attention ->
// out proj + bias + residual.  All GEMM math via v_wmma_f32_16x16x32_f16.
// B operands kept K-contiguous (transposed) so every fragment is 2x b128.
// ---------------------------------------------------------------------------

typedef __attribute__((ext_vector_type(16))) _Float16     v16h;
typedef __attribute__((ext_vector_type(8)))  float        v8f;
typedef __attribute__((ext_vector_type(4)))  unsigned int v4u;

#define DIM      512
#define SEQ      2048
#define BATCH    4
#define HEADS    8
#define DIM_HEAD 64
#define ROWS     (BATCH * SEQ)   // 8192

__device__ __forceinline__ v8f wmma16(v16h a, v16h b, v8f c) {
  return __builtin_amdgcn_wmma_f32_16x16x32_f16(
      /*neg_a=*/false, a, /*neg_b=*/false, b,
      /*c_mod=*/(short)0, c, /*reuse_a=*/false, /*reuse_b=*/false);
}

// A fragment (16x32 f16).  ISA layout: lane L (m=L&15, h=L>>4), VGPR r holds
// K = 2r + (r>=4 ? 8 : 0) + 8h.  Per lane this is two contiguous 16B runs:
// [k_base+8h .. +7] and [k_base+16+8h .. +7]  -> two b128 loads.
__device__ __forceinline__ v16h load_a_frag(const _Float16* A, int lda,
                                            int m_base, int k_base) {
  const int lane = threadIdx.x & 31;
  const int m = lane & 15, hh = lane >> 4;
  const _Float16* base = A + (size_t)(m_base + m) * lda + k_base + 8 * hh;
  union { v4u q[2]; v16h h; } f;
  f.q[0] = *(const v4u*)(base);        // K = 8h .. 8h+7
  f.q[1] = *(const v4u*)(base + 16);   // K = 16+8h .. 16+8h+7
  return f.h;
}

// B fragment (32x16 f16) from a K-contiguous ("transposed") operand
// BT[n][k]: lane L (n=L&15, h=L>>4), same K order as the A fragment:
// [k_base+8h .. +7] and [k_base+16+8h .. +7] -> two b128 loads.
__device__ __forceinline__ v16h load_bT_frag(const _Float16* BT, int ldk,
                                             int n_base, int k_base) {
  const int lane = threadIdx.x & 31;
  const int n = lane & 15, hh = lane >> 4;
  const _Float16* base = BT + (size_t)(n_base + n) * ldk + k_base + 8 * hh;
  union { v4u q[2]; v16h h; } f;
  f.q[0] = *(const v4u*)(base);
  f.q[1] = *(const v4u*)(base + 16);
  return f.h;
}

// ---------------------------------------------------------------------------
// Kernel 1: LayerNorm fp32 -> f16 (one 256-thread block per row of 512)
// ---------------------------------------------------------------------------
__global__ __launch_bounds__(256) void ln_f16_kernel(
    const float* __restrict__ x, const float* __restrict__ g,
    const float* __restrict__ b, _Float16* __restrict__ out) {
  __shared__ float ssum[256];
  __shared__ float ssq[256];
  const int row = blockIdx.x;
  const int t = threadIdx.x;
  const float* xr = x + (size_t)row * DIM;
  float v0 = xr[t];
  float v1 = xr[t + 256];
  ssum[t] = v0 + v1;
  ssq[t]  = v0 * v0 + v1 * v1;
  __syncthreads();
#pragma unroll
  for (int off = 128; off > 0; off >>= 1) {
    if (t < off) { ssum[t] += ssum[t + off]; ssq[t] += ssq[t + off]; }
    __syncthreads();
  }
  const float mean = ssum[0] * (1.0f / DIM);
  const float var  = ssq[0] * (1.0f / DIM) - mean * mean;
  const float rstd = rsqrtf(var + 1e-5f);
  _Float16* orow = out + (size_t)row * DIM;
  orow[t]       = (_Float16)((v0 - mean) * rstd * g[t] + b[t]);
  orow[t + 256] = (_Float16)((v1 - mean) * rstd * g[t + 256] + b[t + 256]);
}

// ---------------------------------------------------------------------------
// Kernel 2: fp32 -> f16 with transpose (weights W[K][N] -> WT[N][K])
// ---------------------------------------------------------------------------
__global__ __launch_bounds__(256) void cvt_f16_t_kernel(
    const float* __restrict__ in, _Float16* __restrict__ outT) {
  const int i = blockIdx.x * blockDim.x + threadIdx.x;  // over DIM*DIM
  const int k = i >> 9;        // row of W
  const int n = i & (DIM - 1); // col of W
  outT[(size_t)n * DIM + k] = (_Float16)in[i];
}

// ---------------------------------------------------------------------------
// Kernel 3: WMMA GEMM  C[M,N] = A[M,K] @ BT[N,K]^T   (f16 in, f16 out)
// 32x64 output per wave (2 A-frags x 4 B-frags -> 8 WMMAs / k-chunk).
// TRANS_OUT=true stores C transposed (CT[N][M]) -- used for V.
// ---------------------------------------------------------------------------
template <bool TRANS_OUT>
__global__ __launch_bounds__(256) void gemm_f16_kernel(
    const _Float16* __restrict__ A, const _Float16* __restrict__ BT,
    _Float16* __restrict__ C, int M, int N, int K) {
  const int wid = blockIdx.x * (blockDim.x >> 5) + (threadIdx.x >> 5);
  const int ntn = N >> 6;                 // 64-wide col blocks
  const int tm = wid / ntn, tn = wid % ntn;
  const int m0 = tm * 32, n0 = tn * 64;
  if (m0 >= M) return;
  const int lane = threadIdx.x & 31;
  const int n16 = lane & 15, hh = lane >> 4;

  v8f acc[2][4];
#pragma unroll
  for (int i = 0; i < 2; ++i)
#pragma unroll
    for (int j = 0; j < 4; ++j) acc[i][j] = (v8f){};

#pragma unroll 2
  for (int kc = 0; kc < K; kc += 32) {
    __builtin_prefetch((const void*)(A + (size_t)m0 * K + kc + 64), 0, 1);
    __builtin_prefetch((const void*)(BT + (size_t)n0 * K + kc + 64), 0, 1);
    v16h a0 = load_a_frag(A, K, m0, kc);
    v16h a1 = load_a_frag(A, K, m0 + 16, kc);
#pragma unroll
    for (int j = 0; j < 4; ++j) {
      v16h b = load_bT_frag(BT, K, n0 + 16 * j, kc);
      acc[0][j] = wmma16(a0, b, acc[0][j]);
      acc[1][j] = wmma16(a1, b, acc[1][j]);
    }
  }

#pragma unroll
  for (int i = 0; i < 2; ++i)
#pragma unroll
    for (int j = 0; j < 4; ++j)
#pragma unroll
      for (int r = 0; r < 8; ++r) {
        const int row = m0 + 16 * i + r + 8 * hh;
        const int col = n0 + 16 * j + n16;
        if (TRANS_OUT)
          C[(size_t)col * M + row] = (_Float16)acc[i][j][r];
        else
          C[(size_t)row * N + col] = (_Float16)acc[i][j][r];
      }
}

// ---------------------------------------------------------------------------
// Kernel 4: flash attention.  One wave = 32 query rows x one head.
// S (32x32 keys) via 8 WMMAs, online softmax, P staged through LDS
// (C/D layout -> A-fragment layout), PV via 8 WMMAs.  V is pre-transposed
// (VT[d][row]) so V fragments are b128 loads.
// ---------------------------------------------------------------------------
__global__ __launch_bounds__(128) void flash_attn_kernel(
    const _Float16* __restrict__ Qp, const _Float16* __restrict__ Kp,
    const _Float16* __restrict__ VT, _Float16* __restrict__ Op,
    float scale) {
  __shared__ __align__(16) _Float16 pbuf[4][2][16 * 32];  // per-wave P tiles
  const int wave = threadIdx.x >> 5;
  const int lane = threadIdx.x & 31;
  const int n16 = lane & 15, hh = lane >> 4;
  const int head = blockIdx.y;
  const int batch = blockIdx.z;
  const int qtile = blockIdx.x * 4 + wave;    // 0..63, 32 rows each
  const int row0 = batch * SEQ + qtile * 32;  // global Q row base
  const int dh0 = head * DIM_HEAD;

  // Q fragments: 2 q-subtiles x 2 k-chunks of dh=64
  v16h aq[2][2];
#pragma unroll
  for (int i = 0; i < 2; ++i) {
    aq[i][0] = load_a_frag(Qp, DIM, row0 + 16 * i, dh0);
    aq[i][1] = load_a_frag(Qp, DIM, row0 + 16 * i, dh0 + 32);
  }

  v8f o[2][4];
  float mrow[2][8], lrow[2][8];
#pragma unroll
  for (int i = 0; i < 2; ++i) {
#pragma unroll
    for (int j = 0; j < 4; ++j) o[i][j] = (v8f){};
#pragma unroll
    for (int r = 0; r < 8; ++r) { mrow[i][r] = -3.0e38f; lrow[i][r] = 0.0f; }
  }

  for (int kb = 0; kb < SEQ; kb += 32) {
    const int krow0 = batch * SEQ + kb;

    // K^T fragments (KT access == bT pattern on row-major K): 2 key halves
    v16h bk[2][2];
#pragma unroll
    for (int half = 0; half < 2; ++half) {
      bk[half][0] = load_bT_frag(Kp, DIM, krow0 + 16 * half, dh0);
      bk[half][1] = load_bT_frag(Kp, DIM, krow0 + 16 * half, dh0 + 32);
    }

    // S = Q @ K^T : 8 WMMAs
    v8f s[2][2];
#pragma unroll
    for (int i = 0; i < 2; ++i)
#pragma unroll
      for (int half = 0; half < 2; ++half) {
        v8f t = (v8f){};
        t = wmma16(aq[i][0], bk[half][0], t);
        t = wmma16(aq[i][1], bk[half][1], t);
        s[i][half] = t;
      }

    // Online softmax per q-subtile (row m -> reg m%8, lane-half m/8)
#pragma unroll
    for (int i = 0; i < 2; ++i) {
      float p0[8], p1[8];
#pragma unroll
      for (int r = 0; r < 8; ++r) {
        float v0 = s[i][0][r] * scale, v1 = s[i][1][r] * scale;
        float mx = fmaxf(v0, v1);
#pragma unroll
        for (int msk = 1; msk < 16; msk <<= 1)
          mx = fmaxf(mx, __shfl_xor(mx, msk, 32));
        const float mnew = fmaxf(mrow[i][r], mx);
        const float alpha = __expf(mrow[i][r] - mnew);
        v0 = __expf(v0 - mnew);
        v1 = __expf(v1 - mnew);
        float rs = v0 + v1;
#pragma unroll
        for (int msk = 1; msk < 16; msk <<= 1)
          rs += __shfl_xor(rs, msk, 32);
        lrow[i][r] = lrow[i][r] * alpha + rs;
        mrow[i][r] = mnew;
#pragma unroll
        for (int j = 0; j < 4; ++j) o[i][j][r] *= alpha;
        p0[r] = v0;
        p1[r] = v1;
      }
      // C/D layout -> row-major LDS [16][32]
      _Float16* pb = pbuf[wave][i];
#pragma unroll
      for (int r = 0; r < 8; ++r) {
        const int rrow = r + 8 * hh;
        pb[rrow * 32 + n16]      = (_Float16)p0[r];
        pb[rrow * 32 + 16 + n16] = (_Float16)p1[r];
      }
    }
    asm volatile("s_wait_dscnt 0" ::: "memory");  // wave-synchronous LDS

    // Re-read P as A fragments (two b128-shaped LDS loads each)
    v16h pa[2];
#pragma unroll
    for (int i = 0; i < 2; ++i) {
      const _Float16* pb = pbuf[wave][i] + n16 * 32 + 8 * hh;
      union { v4u q[2]; v16h h; } f;
      f.q[0] = *(const v4u*)(pb);
      f.q[1] = *(const v4u*)(pb + 16);
      pa[i] = f.h;
    }
    asm volatile("s_wait_dscnt 0" ::: "memory");

    // O += P @ V : V fragments from VT (b128), shared across both q-subtiles
#pragma unroll
    for (int j = 0; j < 4; ++j) {
      v16h bv = load_bT_frag(VT, ROWS, dh0 + 16 * j, krow0);
      o[0][j] = wmma16(pa[0], bv, o[0][j]);
      o[1][j] = wmma16(pa[1], bv, o[1][j]);
    }
  }

  // Normalize and store f16 (row-major [8192][512])
#pragma unroll
  for (int i = 0; i < 2; ++i)
#pragma unroll
    for (int r = 0; r < 8; ++r) {
      const float inv = 1.0f / lrow[i][r];
      const size_t row = (size_t)(row0 + 16 * i + r + 8 * hh);
#pragma unroll
      for (int j = 0; j < 4; ++j)
        Op[row * DIM + dh0 + 16 * j + n16] = (_Float16)(o[i][j][r] * inv);
    }
}

// ---------------------------------------------------------------------------
// Kernel 5: out = attn @ Wo + bo + residual  (fp32 epilogue to d_out)
// Same 32x64-per-wave blocking; BT = Wo^T.
// ---------------------------------------------------------------------------
__global__ __launch_bounds__(256) void gemm_out_kernel(
    const _Float16* __restrict__ A, const _Float16* __restrict__ BT,
    const float* __restrict__ bias, const float* __restrict__ resid,
    float* __restrict__ Out, int M, int N, int K) {
  const int wid = blockIdx.x * (blockDim.x >> 5) + (threadIdx.x >> 5);
  const int ntn = N >> 6;
  const int tm = wid / ntn, tn = wid % ntn;
  const int m0 = tm * 32, n0 = tn * 64;
  if (m0 >= M) return;
  const int lane = threadIdx.x & 31;
  const int n16 = lane & 15, hh = lane >> 4;

  v8f acc[2][4];
#pragma unroll
  for (int i = 0; i < 2; ++i)
#pragma unroll
    for (int j = 0; j < 4; ++j) acc[i][j] = (v8f){};

#pragma unroll 2
  for (int kc = 0; kc < K; kc += 32) {
    __builtin_prefetch((const void*)(A + (size_t)m0 * K + kc + 64), 0, 1);
    __builtin_prefetch((const void*)(BT + (size_t)n0 * K + kc + 64), 0, 1);
    v16h a0 = load_a_frag(A, K, m0, kc);
    v16h a1 = load_a_frag(A, K, m0 + 16, kc);
#pragma unroll
    for (int j = 0; j < 4; ++j) {
      v16h b = load_bT_frag(BT, K, n0 + 16 * j, kc);
      acc[0][j] = wmma16(a0, b, acc[0][j]);
      acc[1][j] = wmma16(a1, b, acc[1][j]);
    }
  }

#pragma unroll
  for (int i = 0; i < 2; ++i)
#pragma unroll
    for (int j = 0; j < 4; ++j) {
      const int col = n0 + 16 * j + n16;
#pragma unroll
      for (int r = 0; r < 8; ++r) {
        const size_t row = (size_t)(m0 + 16 * i + r + 8 * hh);
        Out[row * N + col] = acc[i][j][r] + bias[col] + resid[row * N + col];
      }
    }
}

// ---------------------------------------------------------------------------
// Host launcher
// ---------------------------------------------------------------------------
extern "C" void kernel_launch(void* const* d_in, const int* in_sizes, int n_in,
                              void* d_out, int out_size, void* d_ws, size_t ws_size,
                              hipStream_t stream) {
  (void)in_sizes; (void)n_in; (void)out_size; (void)ws_size;
  const float* x    = (const float*)d_in[0];
  const float* ln_g = (const float*)d_in[1];
  const float* ln_b = (const float*)d_in[2];
  const float* Wq   = (const float*)d_in[3];
  const float* Wk   = (const float*)d_in[4];
  const float* Wv   = (const float*)d_in[5];
  const float* Wo   = (const float*)d_in[6];
  const float* bo   = (const float*)d_in[7];
  float* out = (float*)d_out;

  // Workspace partition (f16 buffers), 256B-aligned slabs
  char* w = (char*)d_ws;
  auto take = [&](size_t bytes) {
    char* p = w;
    w += (bytes + 255) & ~(size_t)255;
    return p;
  };
  _Float16* xn   = (_Float16*)take((size_t)ROWS * DIM * 2);
  _Float16* WqT  = (_Float16*)take((size_t)DIM * DIM * 2);
  _Float16* WkT  = (_Float16*)take((size_t)DIM * DIM * 2);
  _Float16* WvT  = (_Float16*)take((size_t)DIM * DIM * 2);
  _Float16* WoT  = (_Float16*)take((size_t)DIM * DIM * 2);
  _Float16* Qh   = (_Float16*)take((size_t)ROWS * DIM * 2);  // [row][d]
  _Float16* Kh   = (_Float16*)take((size_t)ROWS * DIM * 2);  // [row][d]
  _Float16* VhT  = (_Float16*)take((size_t)ROWS * DIM * 2);  // [d][row]
  _Float16* Ah   = (_Float16*)take((size_t)ROWS * DIM * 2);  // [row][d]

  // Weights fp32 -> f16, transposed to K-contiguous layout
  const int wn = DIM * DIM;
  cvt_f16_t_kernel<<<wn / 256, 256, 0, stream>>>(Wq, WqT);
  cvt_f16_t_kernel<<<wn / 256, 256, 0, stream>>>(Wk, WkT);
  cvt_f16_t_kernel<<<wn / 256, 256, 0, stream>>>(Wv, WvT);
  cvt_f16_t_kernel<<<wn / 256, 256, 0, stream>>>(Wo, WoT);

  // LayerNorm -> f16
  ln_f16_kernel<<<ROWS, 256, 0, stream>>>(x, ln_g, ln_b, xn);

  // Q/K/V projections: (8192/32)*(512/64) = 2048 waves -> 256 blocks
  const int gemm_blocks = (ROWS / 32) * (DIM / 64) / 8;
  gemm_f16_kernel<false><<<gemm_blocks, 256, 0, stream>>>(xn, WqT, Qh,
                                                          ROWS, DIM, DIM);
  gemm_f16_kernel<false><<<gemm_blocks, 256, 0, stream>>>(xn, WkT, Kh,
                                                          ROWS, DIM, DIM);
  gemm_f16_kernel<true><<<gemm_blocks, 256, 0, stream>>>(xn, WvT, VhT,
                                                         ROWS, DIM, DIM);

  // Attention: grid (qtiles32/4, heads, batch), 4 waves per block
  const float scale = 1.0f / sqrtf((float)(HEADS * DIM_HEAD));  // hidden^-0.5
  flash_attn_kernel<<<dim3(SEQ / 32 / 4, HEADS, BATCH), 128, 0, stream>>>(
      Qh, Kh, VhT, Ah, scale);

  // Output projection + bias + residual -> fp32 d_out
  gemm_out_kernel<<<gemm_blocks, 256, 0, stream>>>(Ah, WoT, bo, x, out,
                                                   ROWS, DIM, DIM);
}
